// MambaActor_1529008357588
// MI455X (gfx1250) — matrix operand order repaired and checked by reading, not compiled, not run on hardware
//
#include <hip/hip_runtime.h>
#include <cmath>

typedef __bf16 bf16;
typedef __attribute__((ext_vector_type(16))) __bf16 v16bf;
typedef __attribute__((ext_vector_type(8)))  __bf16 v8bf;
typedef __attribute__((ext_vector_type(8)))  float  v8f;
typedef __attribute__((ext_vector_type(4)))  int    v4i;

#define AS1 __attribute__((address_space(1)))
#define AS3 __attribute__((address_space(3)))

#ifdef __has_builtin
#if __has_builtin(__builtin_amdgcn_global_load_async_to_lds_b128)
#define HAVE_ASYNC_LDS 1
#endif
#if __has_builtin(__builtin_amdgcn_s_wait_asynccnt)
#define HAVE_WAIT_ASYNC 1
#endif
#endif

#define BQ   1024      // batch
#define LQ   32        // seq len
#define CIN  96
#define COUT 12
#define DM   128
#define DIN  256       // D_INNER
#define NH   8         // heads
#define HD   32        // headdim
#define DST  16        // d_state
#define CDIM 288       // conv dim = DIN + 2*DST
#define EIN  552       // 2*DIN + 2*DST + NH
#define MROWS (BQ*LQ)  // 32768

// ---------------------------------------------------------------- utilities
__global__ void k_f2bf(const float* __restrict__ in, bf16* __restrict__ out, int n) {
  int i = blockIdx.x * blockDim.x + threadIdx.x;
  if (i < n) out[i] = (bf16)in[i];
}

// out_buf[b,c,l] = l<31 ? buffer[b,c,l+1] : x[b,c]
__global__ void k_make_buf(const float* __restrict__ x, const float* __restrict__ buffer,
                           float* __restrict__ out_buf, int total) {
  int i = blockIdx.x * blockDim.x + threadIdx.x;
  if (i >= total) return;
  int l = i & (LQ - 1);
  int rest = i >> 5;
  int c = rest % CIN;
  int b = rest / CIN;
  out_buf[i] = (l < LQ - 1) ? buffer[((size_t)b * CIN + c) * LQ + l + 1]
                            : x[(size_t)b * CIN + c];
}

// h[m,d] = sum_c buf[b,c,l] * W[d,c] + bias[d];  m = b*32+l. grid=MROWS, block=128
__global__ void k_fcin(const float* __restrict__ bufq, const float* __restrict__ W,
                       const float* __restrict__ bias, float* __restrict__ h) {
  __shared__ float row[CIN];
  int m = blockIdx.x, t = threadIdx.x;
  int b = m >> 5, l = m & (LQ - 1);
  if (t < CIN) row[t] = bufq[((size_t)b * CIN + t) * LQ + l];
  __syncthreads();
  float s = bias[t];
#pragma unroll 4
  for (int c = 0; c < CIN; ++c) s += row[c] * W[t * CIN + c];
  h[(size_t)m * DM + t] = s;
}

// hh[b, 31-l, :] = h[b, l, :]
__global__ void k_flip(const float* __restrict__ h, float* __restrict__ hh) {
  int i = blockIdx.x * blockDim.x + threadIdx.x;  // MROWS*DM total
  if (i >= MROWS * DM) return;
  int d = i & (DM - 1);
  int m = i >> 7;
  int b = m >> 5, l = m & (LQ - 1);
  hh[(((size_t)b * LQ) + (LQ - 1 - l)) * DM + d] = h[i];
}

// rmsnorm over `cols` (block = cols), bf16 output. grid=rows
__global__ void k_rmsnorm_bf16(const float* __restrict__ in, const float* __restrict__ w,
                               bf16* __restrict__ out, int cols) {
  extern __shared__ float red[];
  int m = blockIdx.x, t = threadIdx.x;
  float v = in[(size_t)m * cols + t];
  red[t] = v * v;
  __syncthreads();
  for (int s = blockDim.x >> 1; s > 0; s >>= 1) {
    if (t < s) red[t] += red[t + s];
    __syncthreads();
  }
  float rs = rsqrtf(red[0] / (float)cols + 1e-5f);
  out[(size_t)m * cols + t] = (bf16)(v * rs * w[t]);
}

// final per-direction rmsnorm (cols=128) accumulated into hs with optional flip
__global__ void k_final_norm(const float* __restrict__ in, const float* __restrict__ w,
                             float* __restrict__ hs, int dir) {
  __shared__ float red[DM];
  int m = blockIdx.x, t = threadIdx.x;
  int b = m >> 5, l = m & (LQ - 1);
  float v = in[(size_t)m * DM + t];
  red[t] = v * v;
  __syncthreads();
  for (int s = DM >> 1; s > 0; s >>= 1) {
    if (t < s) red[t] += red[t + s];
    __syncthreads();
  }
  float rs = rsqrtf(red[0] / (float)DM + 1e-5f);
  float val = v * rs * w[t];
  int lo = dir ? (LQ - 1 - l) : l;
  size_t o = ((size_t)b * LQ + lo) * DM + t;
  if (dir) hs[o] += val; else hs[o] = val;
}

// ------------------------------------------------------------ WMMA bf16 GEMM
// Out[M,N] = A[M,K](bf16) x W[N,K]^T(bf16)  (+ Res[M,N] if Res != null)
// Block = 256 thr (8 waves). Block tile = 128 rows x 64 cols:
//   wave w -> M-tile (mg*8+w); each wave computes 4 N-tiles (one A frag, 4 WMMA).
// Weight slice [64 x K] staged in LDS via async global->LDS DMA (ASYNCcnt).
// K multiple of 32; M multiple of 128; N guarded at the right edge.
__global__ void k_gemm_bf16(const bf16* __restrict__ A, const bf16* __restrict__ W,
                            float* __restrict__ Out, const float* __restrict__ Res,
                            int M, int N, int K, int ngroups) {
  extern __shared__ bf16 Wlds[];           // 64 * K halves
  int tid = threadIdx.x;
  int mg  = blockIdx.x / ngroups;          // 128-row group
  int ng  = blockIdx.x - mg * ngroups;     // 64-col group
  int n0  = ng * 64;

  // ---- stage weight slice [64 x K] into LDS (async DMA, 16B chunks) ----
  int chunks = (64 * K) >> 3;              // # of 8-half (16B) chunks
  for (int idx = tid; idx < chunks; idx += 256) {
    int halfidx = idx << 3;
    int cl = halfidx / K;                  // local col 0..63 (K is pow2)
    int kk = halfidx - cl * K;
    int cg = n0 + cl; if (cg > N - 1) cg = N - 1;   // clamp right edge
    const bf16* src = W + (size_t)cg * K + kk;
    bf16* dst = Wlds + halfidx;
#ifdef HAVE_ASYNC_LDS
    __builtin_amdgcn_global_load_async_to_lds_b128((AS1 v4i*)src, (AS3 v4i*)dst, 0, 0);
#else
    *(v8bf*)dst = *(const v8bf*)src;
#endif
  }
#ifdef HAVE_ASYNC_LDS
#ifdef HAVE_WAIT_ASYNC
  __builtin_amdgcn_s_wait_asynccnt(0);
#else
  asm volatile("s_wait_asynccnt 0" ::: "memory");
#endif
#endif
  __syncthreads();

  int lane = tid & 31;
  int hi   = lane >> 4;                    // lane half: 0 or 1
  int mt   = mg * 8 + (tid >> 5);          // this wave's M-tile
  int rowa = mt * 16 + (lane & 15);        // A row for this lane

  v8f acc[4];
#pragma unroll
  for (int j = 0; j < 4; ++j) {
    int col = n0 + j * 16 + (lane & 15);
    if (Res && col < N) {
#pragma unroll
      for (int r = 0; r < 8; ++r)
        acc[j][r] = Res[(size_t)(mt * 16 + r + hi * 8) * N + col];
    } else {
#pragma unroll
      for (int r = 0; r < 8; ++r) acc[j][r] = 0.f;
    }
  }

  const bf16* arow = A + (size_t)rowa * K;
  for (int k0 = 0; k0 < K; k0 += 32) {
    // prefetch next K-slice of A into caches (global_prefetch_b8)
    int kpf = (k0 + 32 < K) ? (k0 + 32) : k0;
    __builtin_prefetch(arow + kpf, 0, 0);
    // A fragment: halves 0-15 = K[k0 + hi*16 ..] (contiguous)
    const v8bf a0 = *(const v8bf*)(arow + k0 + hi * 16);
    const v8bf a1 = *(const v8bf*)(arow + k0 + hi * 16 + 8);
    v16bf a;
#pragma unroll
    for (int j = 0; j < 8; ++j) { a[j] = a0[j]; a[j + 8] = a1[j]; }
    // 4 WMMAs reuse this A fragment; B fragments from LDS
#pragma unroll
    for (int j = 0; j < 4; ++j) {
      int cl = j * 16 + (lane & 15);
      v16bf bfrag = *(const v16bf*)(Wlds + (size_t)cl * K + k0 + hi * 16);
      acc[j] = __builtin_amdgcn_wmma_f32_16x16x32_bf16(false, a, false, bfrag,
                                                       (short)0, acc[j], false, false);
    }
  }

#pragma unroll
  for (int j = 0; j < 4; ++j) {
    int col = n0 + j * 16 + (lane & 15);
    if (col < N) {
#pragma unroll
      for (int r = 0; r < 8; ++r)
        Out[(size_t)(mt * 16 + r + hi * 8) * N + col] = acc[j][r];
    }
  }
}

// ----------------------------------------------- depthwise causal conv + silu
__global__ void k_conv(const float* __restrict__ zx, const float* __restrict__ cw,
                       const float* __restrict__ cb, float* __restrict__ xBCc) {
  int i = blockIdx.x * blockDim.x + threadIdx.x;
  if (i >= BQ * CDIM) return;
  int c = i % CDIM, b = i / CDIM;
  float w0 = cw[c * 4 + 0], w1 = cw[c * 4 + 1], w2 = cw[c * 4 + 2], w3 = cw[c * 4 + 3];
  float bb = cb[c];
  float x0 = 0.f, x1 = 0.f, x2 = 0.f;
  for (int l = 0; l < LQ; ++l) {
    size_t m = (size_t)b * LQ + l;
    float xl = zx[m * EIN + DIN + c];
    float a = bb + w0 * x0 + w1 * x1 + w2 * x2 + w3 * xl;
    a = a / (1.f + __expf(-a));
    xBCc[m * CDIM + c] = a;
    x0 = x1; x1 = x2; x2 = xl;
  }
}

// --------------------------------------------------------------- SSM scan
// grid = B, block = 256 (thread = head*32 + p). 16-state recurrence in regs.
__global__ void k_scan(const float* __restrict__ zx, const float* __restrict__ xBCc,
                       const float* __restrict__ dtb, const float* __restrict__ alog,
                       const float* __restrict__ Dp, float* __restrict__ yg) {
  __shared__ float sB[LQ * DST], sC[LQ * DST], sdt[NH * LQ], sdA[NH * LQ];
  int b = blockIdx.x, t = threadIdx.x;
  for (int idx = t; idx < LQ * DST; idx += 256) {
    int l = idx >> 4, n = idx & 15;
    size_t m = (size_t)b * LQ + l;
    sB[idx] = xBCc[m * CDIM + DIN + n];
    sC[idx] = xBCc[m * CDIM + DIN + DST + n];
  }
  {
    int h = t >> 5, l = t & 31;
    size_t m = (size_t)b * LQ + l;
    float raw = zx[m * EIN + DIN + CDIM + h] + dtb[h];
    float sp = (raw > 20.f) ? raw : log1pf(__expf(raw));
    sdt[t] = sp;
    sdA[t] = __expf(-__expf(alog[h]) * sp);
  }
  __syncthreads();
  int h = t >> 5, p = t & 31;
  float st[DST];
#pragma unroll
  for (int n = 0; n < DST; ++n) st[n] = 0.f;
  float Dh = Dp[h];
  for (int l = 0; l < LQ; ++l) {
    size_t m = (size_t)b * LQ + l;
    float xs  = xBCc[m * CDIM + h * HD + p];
    float a   = sdA[h * LQ + l];
    float dtx = sdt[h * LQ + l] * xs;
    float y = 0.f;
#pragma unroll
    for (int n = 0; n < DST; ++n) {
      st[n] = st[n] * a + dtx * sB[l * DST + n];
      y += st[n] * sC[l * DST + n];
    }
    y += Dh * xs;
    float z = zx[m * EIN + h * HD + p];
    yg[m * DIN + h * HD + p] = y * (z / (1.f + __expf(-z)));
  }
}

// -------------------------------------------------------------- output head
__global__ void k_head(const float* __restrict__ hs, const float* __restrict__ fw,
                       const float* __restrict__ fb, const float* __restrict__ lw,
                       const float* __restrict__ lb, float* __restrict__ y) {
  __shared__ float row[LQ * DM];
  int b = blockIdx.x, t = threadIdx.x;  // block = 128
  for (int i = t; i < LQ * DM; i += 128) row[i] = hs[(size_t)b * LQ * DM + i];
  __syncthreads();
  if (t < COUT) {
    float acc = lb[0];
    for (int l = 0; l < LQ; ++l) {
      float d = fb[t];
#pragma unroll 4
      for (int k = 0; k < DM; ++k) d += row[l * DM + k] * fw[t * DM + k];
      acc += d * lw[l];
    }
    y[(size_t)b * COUT + t] = acc;
  }
}

// ------------------------------------------------------------------- launch
extern "C" void kernel_launch(void* const* d_in, const int* in_sizes, int n_in,
                              void* d_out, int out_size, void* d_ws, size_t ws_size,
                              hipStream_t stream) {
  const float* x            = (const float*)d_in[0];
  const float* buffer       = (const float*)d_in[1];
  const float* fc_in_w      = (const float*)d_in[2];
  const float* fc_in_b      = (const float*)d_in[3];
  const float* blk_norm_w   = (const float*)d_in[4];   // 4 x 128
  const float* in_proj_w    = (const float*)d_in[5];   // 4 x 552 x 128
  const float* conv_w       = (const float*)d_in[6];   // 4 x 288 x 4
  const float* conv_b       = (const float*)d_in[7];   // 4 x 288
  const float* dt_bias      = (const float*)d_in[8];   // 4 x 8
  const float* A_log        = (const float*)d_in[9];   // 4 x 8
  const float* Dski         = (const float*)d_in[10];  // 4 x 8
  const float* mixer_norm_w = (const float*)d_in[11];  // 4 x 256
  const float* out_proj_w   = (const float*)d_in[12];  // 4 x 128 x 256
  const float* norm_f_w     = (const float*)d_in[13];  // 2 x 128
  const float* fc_out_w     = (const float*)d_in[14];  // 12 x 128
  const float* fc_out_b     = (const float*)d_in[15];
  const float* out_lin_w    = (const float*)d_in[16];  // 1 x 32
  const float* out_lin_b    = (const float*)d_in[17];

  float* out_y   = (float*)d_out;                 // 1024*12
  float* out_buf = out_y + (size_t)BQ * COUT;     // 1024*96*32

  // workspace carve-out (256B aligned)
  char* wsp = (char*)d_ws;
  auto carve = [&](size_t bytes) { char* p = wsp; wsp += (bytes + 255) & ~(size_t)255; return p; };
  float* h    = (float*)carve((size_t)MROWS * DM * 4);
  float* hh   = (float*)carve((size_t)MROWS * DM * 4);
  float* hs   = (float*)carve((size_t)MROWS * DM * 4);
  float* zx   = (float*)carve((size_t)MROWS * EIN * 4);
  float* xBCc = (float*)carve((size_t)MROWS * CDIM * 4);
  float* yg   = (float*)carve((size_t)MROWS * DIN * 4);
  bf16* hn    = (bf16*)carve((size_t)MROWS * DM * 2);
  bf16* gn    = (bf16*)carve((size_t)MROWS * DIN * 2);
  bf16* wpin  = (bf16*)carve((size_t)4 * EIN * DM * 2);
  bf16* wpout = (bf16*)carve((size_t)4 * DM * DIN * 2);

  // buffer shift + weight conversion + fc_in
  {
    int total = BQ * CIN * LQ;
    k_make_buf<<<(total + 255) / 256, 256, 0, stream>>>(x, buffer, out_buf, total);
  }
  {
    int n1 = 4 * EIN * DM, n2 = 4 * DM * DIN;
    k_f2bf<<<(n1 + 255) / 256, 256, 0, stream>>>(in_proj_w, wpin, n1);
    k_f2bf<<<(n2 + 255) / 256, 256, 0, stream>>>(out_proj_w, wpout, n2);
  }
  k_fcin<<<MROWS, DM, 0, stream>>>(out_buf, fc_in_w, fc_in_b, h);

  for (int dir = 0; dir < 2; ++dir) {
    if (dir == 0) {
      (void)hipMemcpyAsync(hh, h, (size_t)MROWS * DM * 4, hipMemcpyDeviceToDevice, stream);
    } else {
      int total = MROWS * DM;
      k_flip<<<(total + 255) / 256, 256, 0, stream>>>(h, hh);
    }
    for (int l2 = 0; l2 < 2; ++l2) {
      int i = dir * 2 + l2;
      // pre-norm
      k_rmsnorm_bf16<<<MROWS, DM, DM * 4, stream>>>(hh, blk_norm_w + i * DM, hn, DM);
      // in_proj: [32768,128] x [128,552]^T  (WMMA, LDS-staged weights)
      {
        int ngroups = (EIN + 63) / 64;              // 9
        int blocks  = (MROWS / 128) * ngroups;      // 2304
        k_gemm_bf16<<<blocks, 256, 64 * DM * 2, stream>>>(
            hn, wpin + (size_t)i * EIN * DM, zx, nullptr, MROWS, EIN, DM, ngroups);
      }
      // depthwise conv + silu
      {
        int total = BQ * CDIM;
        k_conv<<<(total + 255) / 256, 256, 0, stream>>>(zx, conv_w + i * CDIM * 4,
                                                        conv_b + i * CDIM, xBCc);
      }
      // SSM scan + D skip + silu(z) gating
      k_scan<<<BQ, 256, 0, stream>>>(zx, xBCc, dt_bias + i * NH, A_log + i * NH,
                                     Dski + i * NH, yg);
      // mixer rmsnorm (256)
      k_rmsnorm_bf16<<<MROWS, DIN, DIN * 4, stream>>>(yg, mixer_norm_w + i * DIN, gn, DIN);
      // out_proj + residual: hh += [32768,256] x [256,128]^T  (WMMA)
      {
        int ngroups = DM / 64;                      // 2
        int blocks  = (MROWS / 128) * ngroups;      // 512
        k_gemm_bf16<<<blocks, 256, 64 * DIN * 2, stream>>>(
            gn, wpout + (size_t)i * DM * DIN, hh, hh, MROWS, DM, DIN, ngroups);
      }
    }
    // final norm for this direction, accumulate (flipped for dir 1) into hs
    k_final_norm<<<MROWS, DM, 0, stream>>>(hh, norm_f_w + dir * DM, hs, dir);
  }

  // output head
  k_head<<<BQ, 128, 0, stream>>>(hs, fc_out_w, fc_out_b, out_lin_w, out_lin_b, out_y);
}
